// LSTM_encoder_13185549598725
// MI455X (gfx1250) — matrix operand order repaired and checked by reading, not compiled, not run on hardware
//
#include <hip/hip_runtime.h>
#include <hip/hip_bf16.h>

// ---------------- problem constants ----------------
#define B_    256
#define S_    50
#define C_    40
#define D_    256
#define V_    20000
#define TDIFF 1000
#define BS_   (B_ * S_)          // 12800

typedef __bf16 bf16_t;
typedef __attribute__((ext_vector_type(16))) __bf16     v16bf;
typedef __attribute__((ext_vector_type(8)))  float      v8f;
typedef __attribute__((ext_vector_type(4)))  unsigned   uv4;   // POD 16-byte vector

// ---- gfx1250 async global->LDS copy (ASYNCcnt-tracked, no VGPR round trip) ----
// LDS byte address = low 32 bits of the generic pointer.
__device__ __forceinline__ void async_ld16(void* lds, const void* g)
{
    unsigned loff = (unsigned)(unsigned long long)lds;
    asm volatile("global_load_async_to_lds_b128 %0, %1, off"
                 :: "v"(loff), "v"(g) : "memory");
}
__device__ __forceinline__ void wait_async_le4()
{
    asm volatile("s_wait_asynccnt 0x4" ::: "memory");
}
__device__ __forceinline__ void wait_async_0()
{
    asm volatile("s_wait_asynccnt 0x0" ::: "memory");
}

// stage a 128x32 bf16 tile (row stride ldr elements) into LDS, 2 x 16B per thread
__device__ __forceinline__ void stage_tile(bf16_t* sT, const bf16_t* src, int ldr, int tid)
{
    int s0 = tid, s1 = tid + 256;
    async_ld16(&sT[(s0 >> 2) * 32 + (s0 & 3) * 8],
               src + (size_t)(s0 >> 2) * ldr + (s0 & 3) * 8);
    async_ld16(&sT[(s1 >> 2) * 32 + (s1 & 3) * 8],
               src + (size_t)(s1 >> 2) * ldr + (s1 & 3) * 8);
}

// =====================================================================
// bf16 WMMA GEMM:  C[M,N] = A[M,K] @ B[K,N] + bias[N],  with B given
// TRANSPOSED (Bn is N x K row-major).  A may be split at k=256:
//   A(m,k) = (k < 256 || A1==null) ? A0[m*lda0+k] : A1[m*lda1+k-256]
// Block tile 128x128, BK=32, 256 threads (8 waves), each wave 64x32
// (4x2 tiles of v_wmma_f32_16x16x32_bf16).  Double-buffered async LDS.
// M,N multiples of 128; K multiple of 32 (true for all uses here).
// =====================================================================
__global__ __launch_bounds__(256)
void gemm_bf16_wmma(const bf16_t* __restrict__ A0, int lda0,
                    const bf16_t* __restrict__ A1, int lda1,
                    const bf16_t* __restrict__ Bn, int ldb,   // N x K row-major
                    const float*  __restrict__ bias,          // length N (may be null)
                    float* __restrict__ C, int ldc,
                    int M, int N, int K)
{
    __shared__ __align__(16) bf16_t sA[2][128 * 32];   // [m][k]
    __shared__ __align__(16) bf16_t sB[2][128 * 32];   // [n][k]

    const int tid  = threadIdx.x;
    const int lane = tid & 31;
    const int wv   = tid >> 5;
    const int wr   = wv >> 2;          // 0..1  (M direction)
    const int wc   = wv & 3;           // 0..3  (N direction)
    const int m0   = blockIdx.x * 128;
    const int n0   = blockIdx.y * 128;
    const int nchunks = K >> 5;

    v8f zero = {};
    v8f acc[4][2];
#pragma unroll
    for (int i = 0; i < 4; i++)
#pragma unroll
        for (int j = 0; j < 2; j++) acc[i][j] = zero;

    // prologue: stage chunk 0
    {
        const bf16_t* Asrc = A0; int lda = lda0;
        stage_tile(sA[0], Asrc + (size_t)m0 * lda, lda, tid);
        stage_tile(sB[0], Bn + (size_t)n0 * ldb, ldb, tid);
    }

    for (int c = 0; c < nchunks; c++) {
        const int cur = c & 1;
        if (c + 1 < nchunks) {
            const int k1 = (c + 1) * 32;
            const bf16_t* Asrc; int lda; int kk;
            if (k1 < 256 || A1 == nullptr) { Asrc = A0; lda = lda0; kk = k1; }
            else                           { Asrc = A1; lda = lda1; kk = k1 - 256; }
            stage_tile(sA[cur ^ 1], Asrc + (size_t)m0 * lda + kk, lda, tid);
            stage_tile(sB[cur ^ 1], Bn + (size_t)n0 * ldb + k1, ldb, tid);
            wait_async_le4();           // chunk c complete; chunk c+1 in flight
        } else {
            wait_async_0();
        }
        __syncthreads();

        const bf16_t* cA = sA[cur];
        const bf16_t* cB = sB[cur];

        // ---- B fragments: col=lane&15, elems 0..7 = K[kb..], 8..15 = K[16+kb..], kb = (lane>>4)*8 ----
        const int kb = (lane < 16) ? 0 : 8;
        v16bf bfr[2];
#pragma unroll
        for (int ns = 0; ns < 2; ns++) {
            int nn = wc * 32 + ns * 16 + (lane & 15);
            union { uv4 u[2]; v16bf v; } tb;
            tb.u[0] = *(const uv4*)(&cB[nn * 32 + kb]);
            tb.u[1] = *(const uv4*)(&cB[nn * 32 + kb + 16]);
            bfr[ns] = tb.v;
        }
        // ---- A fragments: row=lane&15, elems 0..7 = K[kg*8..], 8..15 = K[16+kg*8..] ----
        const int kg = lane >> 4;
#pragma unroll
        for (int ms = 0; ms < 4; ms++) {
            int mm = wr * 64 + ms * 16 + (lane & 15);
            union { uv4 u[2]; v16bf v; } ta;
            ta.u[0] = *(const uv4*)(&cA[mm * 32 + kg * 8]);
            ta.u[1] = *(const uv4*)(&cA[mm * 32 + 16 + kg * 8]);
            v16bf afr = ta.v;
#pragma unroll
            for (int ns = 0; ns < 2; ns++) {
                acc[ms][ns] = __builtin_amdgcn_wmma_f32_16x16x32_bf16(
                    false, afr, false, bfr[ns], (short)0, acc[ms][ns], false, false);
            }
        }
        __syncthreads();   // all waves done reading buffers before next overwrite
    }

    // ---- epilogue: C/D layout: VGPR r -> M = base + (lane<16 ? r : 8+r), N = lane&15 ----
#pragma unroll
    for (int ms = 0; ms < 4; ms++) {
#pragma unroll
        for (int ns = 0; ns < 2; ns++) {
            int nn = n0 + wc * 32 + ns * 16 + (lane & 15);
            float bv = bias ? bias[nn] : 0.0f;
            int mbase = m0 + wr * 64 + ms * 16 + ((lane < 16) ? 0 : 8);
#pragma unroll
            for (int r = 0; r < 8; r++) {
                C[(size_t)(mbase + r) * ldc + nn] = acc[ms][ns][r] + bv;
            }
        }
    }
}

// =====================================================================
// Weight prep: bf16 casts into N-major (already-transposed) layouts.
//   WcatN[n][k] (1024 x 512) = k<256 ? Wih[n][k] : Whh[n][k-256]
//   WhkN [n][k] (256 x 256)  = Whk[n][k]          (natural layout)
//   WdN  [n][k] (256 x 256)  = Wdiff[k][n]        (real transpose)
// =====================================================================
__global__ __launch_bounds__(256)
void prep_weights_kernel(const float* __restrict__ Wih, const float* __restrict__ Whh,
                         const float* __restrict__ bih, const float* __restrict__ bhh,
                         const float* __restrict__ Whk, const float* __restrict__ Wdiff,
                         bf16_t* __restrict__ WcatN, bf16_t* __restrict__ WhkN,
                         bf16_t* __restrict__ WdN, float* __restrict__ gb)
{
    int i = blockIdx.x * 256 + threadIdx.x;            // up to 524288
    if (i < 512 * 1024) {
        int n = i >> 9, k = i & 511;
        float v = (k < 256) ? Wih[(size_t)n * 256 + k] : Whh[(size_t)n * 256 + (k - 256)];
        WcatN[i] = (bf16_t)v;
    }
    if (i < 65536) {
        WhkN[i] = (bf16_t)Whk[i];
        int k = i >> 8, n = i & 255;
        WdN[(size_t)n * 256 + k] = (bf16_t)Wdiff[i];
    }
    if (i < 1024) gb[i] = bih[i] + bhh[i];
}

__global__ void alpha_kernel(float* __restrict__ alpha)
{
    if (blockIdx.x == 0 && threadIdx.x == 0) {
        float a = 1.0f;
        for (int i = 0; i < TDIFF; i++) {
            float beta = 1e-4f + (0.02f - 1e-4f) * (float)i / (float)(TDIFF - 1);
            a *= (1.0f - beta);
            alpha[i] = a;
        }
    }
}

// x[b,s,:] = sum_c emb[idx[b,s,c], :]
__global__ __launch_bounds__(256)
void gather_sum_kernel(const int* __restrict__ idx, const float* __restrict__ emb,
                       float* __restrict__ x, bf16_t* __restrict__ xb)
{
    __shared__ int sidx[C_];
    int bs = blockIdx.x;
    int d  = threadIdx.x;
    if (d < C_) sidx[d] = idx[(size_t)bs * C_ + d];
    __syncthreads();
    float acc = 0.0f;
#pragma unroll 4
    for (int c = 0; c < C_; c++) acc += emb[(size_t)sidx[c] * D_ + d];
    x [(size_t)bs * D_ + d] = acc;
    xb[(size_t)bs * D_ + d] = (bf16_t)acc;
}

// gate activations + c/h update for one timestep
__global__ __launch_bounds__(256)
void lstm_update_kernel(const float* __restrict__ g, float* __restrict__ cstate,
                        bf16_t* __restrict__ rnnb, int t)
{
    int b = blockIdx.x, d = threadIdx.x;
    const float* gr = g + (size_t)b * 1024;
    float i  = 1.0f / (1.0f + expf(-gr[d]));
    float f  = 1.0f / (1.0f + expf(-gr[256 + d]));
    float gg = tanhf(gr[512 + d]);
    float o  = 1.0f / (1.0f + expf(-gr[768 + d]));
    float cp = t ? cstate[b * 256 + d] : 0.0f;
    float c  = f * cp + i * gg;
    cstate[b * 256 + d] = c;
    float h = o * tanhf(c);
    rnnb[((size_t)b * S_ + t) * D_ + d] = (bf16_t)h;
}

// attention + aligned_e_t (row s==0 copies e_k; s>=1 uses attn over [e_k, w[s-1]])
__global__ __launch_bounds__(256)
void attn_align_kernel(const float* __restrict__ x, const float* __restrict__ w,
                       const float* __restrict__ W1, const float* __restrict__ b1,
                       const float* __restrict__ W2, const float* __restrict__ b2,
                       float* __restrict__ aligned)
{
    __shared__ float se[256], sw[256], sh[64], sa[2];
    int bs = blockIdx.x;
    int b = bs / S_, s = bs % S_;
    int d = threadIdx.x;
    float ek = x[(size_t)b * S_ * D_ + d];            // x[b,0,d]
    if (s == 0) { aligned[(size_t)bs * D_ + d] = ek; return; }
    float wp = w[((size_t)b * S_ + (s - 1)) * D_ + d];
    se[d] = ek; sw[d] = wp;
    __syncthreads();
    if (d < 64) {
        float acc = b1[d];
        const float* wr = W1 + (size_t)d * 512;
        for (int k = 0; k < 256; k++) acc += se[k] * wr[k];
        for (int k = 0; k < 256; k++) acc += sw[k] * wr[256 + k];
        sh[d] = tanhf(acc);
    }
    __syncthreads();
    if (d == 0) {
        float l0 = b2[0], l1 = b2[1];
        for (int j = 0; j < 64; j++) { l0 += sh[j] * W2[j]; l1 += sh[j] * W2[64 + j]; }
        float mx = fmaxf(l0, l1);
        float e0 = expf(l0 - mx), e1 = expf(l1 - mx);
        float inv = 1.0f / (e0 + e1);
        sa[0] = e0 * inv; sa[1] = e1 * inv;
    }
    __syncthreads();
    aligned[(size_t)bs * D_ + d] = ek * sa[0] + wp * sa[1];
}

// noisy = aligned*sqrt(a) + noise*sqrt(1-a) + temb[t]   (bf16 for Wdiff GEMM)
__global__ __launch_bounds__(256)
void noisy_kernel(const float* __restrict__ aligned, const float* __restrict__ noise,
                  const int* __restrict__ tdiff, const float* __restrict__ alpha,
                  const float* __restrict__ temb, bf16_t* __restrict__ noisyb)
{
    int m = blockIdx.x;
    int b = m / S_;
    int d = threadIdx.x;
    int t = tdiff[b];
    float al = alpha[t];
    float sa = sqrtf(al), sb = sqrtf(1.0f - al);
    float v = aligned[(size_t)m * D_ + d] * sa + noise[(size_t)m * D_ + d] * sb
            + temb[(size_t)t * D_ + d];
    noisyb[(size_t)m * D_ + d] = (bf16_t)v;
}

// max-pool over s then 2-way projection, for both x and gen_x (computed on the fly)
__global__ __launch_bounds__(256)
void pool_out_kernel(const float* __restrict__ x, const float* __restrict__ aligned,
                     const float* __restrict__ noise, const float* __restrict__ pred,
                     const float* __restrict__ Wout, const float* __restrict__ bout,
                     float* __restrict__ out_pool, float* __restrict__ out_gpool)
{
    __shared__ float smx[256], smg[256];
    int b = blockIdx.x, d = threadIdx.x;
    float mx = -3.0e38f, mg = -3.0e38f;
    for (int s = 0; s < S_; s++) {
        size_t m = (size_t)b * S_ + s;
        float xv = x[m * D_ + d];
        mx = fmaxf(mx, xv);
        float gv = aligned[m * D_ + d] + noise[m * D_ + d] - pred[m * D_ + d];
        mg = fmaxf(mg, gv);
    }
    smx[d] = mx; smg[d] = mg;
    __syncthreads();
    if (d < 2) {
        float a = bout[d], g = bout[d];
        const float* wr = Wout + (size_t)d * D_;
        for (int k = 0; k < 256; k++) { a += smx[k] * wr[k]; g += smg[k] * wr[k]; }
        out_pool [b * 2 + d] = a;
        out_gpool[b * 2 + d] = g;
    }
}

// =====================================================================
extern "C" void kernel_launch(void* const* d_in, const int* in_sizes, int n_in,
                              void* d_out, int out_size, void* d_ws, size_t ws_size,
                              hipStream_t stream)
{
    const int*   input_seqs = (const int*)  d_in[0];
    const int*   tdiff      = (const int*)  d_in[5];
    const float* noise      = (const float*)d_in[6];
    const float* emb        = (const float*)d_in[7];
    const float* Wih        = (const float*)d_in[8];
    const float* Whh        = (const float*)d_in[9];
    const float* bih        = (const float*)d_in[10];
    const float* bhh        = (const float*)d_in[11];
    const float* Whk        = (const float*)d_in[12];
    const float* bhk        = (const float*)d_in[13];
    const float* W1         = (const float*)d_in[14];
    const float* b1         = (const float*)d_in[15];
    const float* W2         = (const float*)d_in[16];
    const float* b2         = (const float*)d_in[17];
    const float* Wdiff      = (const float*)d_in[18];
    const float* bdiff      = (const float*)d_in[19];
    const float* temb       = (const float*)d_in[20];
    const float* Wout       = (const float*)d_in[21];
    const float* bout       = (const float*)d_in[22];

    // workspace carve-out (all buffers fully overwritten every call)
    char* p = (char*)d_ws;
    auto alloc = [&](size_t bytes) -> char* {
        char* r = p; p += (bytes + 255) & ~(size_t)255; return r;
    };
    float*  alpha = (float*) alloc(TDIFF * 4);
    float*  gb    = (float*) alloc(1024 * 4);
    bf16_t* WcatN = (bf16_t*)alloc((size_t)1024 * 512 * 2);
    bf16_t* WhkN  = (bf16_t*)alloc((size_t)256 * 256 * 2);
    bf16_t* WdN   = (bf16_t*)alloc((size_t)256 * 256 * 2);
    float*  x     = (float*) alloc((size_t)BS_ * D_ * 4);
    bf16_t* xb    = (bf16_t*)alloc((size_t)BS_ * D_ * 2);
    bf16_t* rnnb  = (bf16_t*)alloc((size_t)BS_ * D_ * 2);
    float*  cst   = (float*) alloc((size_t)B_ * D_ * 4);
    float*  gbuf  = (float*) alloc((size_t)B_ * 1024 * 4);
    float*  wbuf  = (float*) alloc((size_t)BS_ * D_ * 4);
    float*  algn  = (float*) alloc((size_t)BS_ * D_ * 4);
    bf16_t* nzb   = (bf16_t*)alloc((size_t)BS_ * D_ * 2);

    float* out        = (float*)d_out;
    float* out_pool   = out;
    float* out_gpool  = out + 512;
    float* out_pred   = out + 1024;
    float* out_noise  = out + 1024 + (size_t)BS_ * D_;

    prep_weights_kernel<<<2048, 256, 0, stream>>>(Wih, Whh, bih, bhh, Whk, Wdiff,
                                                  WcatN, WhkN, WdN, gb);
    alpha_kernel<<<1, 64, 0, stream>>>(alpha);
    gather_sum_kernel<<<BS_, 256, 0, stream>>>(input_seqs, emb, x, xb);

    // LSTM: per step, fused [x_t ; h_{t-1}] (256x512) @ Wcat^T + gb
    for (int t = 0; t < S_; t++) {
        const bf16_t* A0 = xb + (size_t)t * D_;
        const bf16_t* A1 = t ? (rnnb + (size_t)(t - 1) * D_) : nullptr;
        gemm_bf16_wmma<<<dim3(2, 8), 256, 0, stream>>>(
            A0, S_ * D_, A1, S_ * D_, WcatN, 512, gb, gbuf, 1024,
            B_, 1024, t ? 512 : 256);
        lstm_update_kernel<<<B_, 256, 0, stream>>>(gbuf, cst, rnnb, t);
    }

    // w = rnn_out @ Whk^T + bhk
    gemm_bf16_wmma<<<dim3(100, 2), 256, 0, stream>>>(
        rnnb, D_, nullptr, 0, WhkN, D_, bhk, wbuf, D_, BS_, D_, D_);

    attn_align_kernel<<<BS_, 256, 0, stream>>>(x, wbuf, W1, b1, W2, b2, algn);
    noisy_kernel<<<BS_, 256, 0, stream>>>(algn, noise, tdiff, alpha, temb, nzb);

    // predicted_noise = (noisy + temb) @ Wdiff + bdiff  -> directly into d_out
    gemm_bf16_wmma<<<dim3(100, 2), 256, 0, stream>>>(
        nzb, D_, nullptr, 0, WdN, D_, bdiff, out_pred, D_, BS_, D_, D_);

    pool_out_kernel<<<B_, 256, 0, stream>>>(x, algn, noise, out_pred, Wout, bout,
                                            out_pool, out_gpool);

    hipMemcpyAsync(out_noise, noise, (size_t)BS_ * D_ * 4,
                   hipMemcpyDeviceToDevice, stream);
}
